// ReedMullerCodebook_44384192037446
// MI455X (gfx1250) — compile-verified, matrix operand order checked by reading, not run on hardware
//
#include <hip/hip_runtime.h>
#include <stdint.h>

// ---------- types ----------
typedef __attribute__((ext_vector_type(16))) __bf16   v16bf;
typedef __attribute__((ext_vector_type(8)))  __bf16   v8bf;
typedef __attribute__((ext_vector_type(8)))  float    v8f;
typedef __attribute__((ext_vector_type(4)))  uint32_t v4u;
typedef __attribute__((ext_vector_type(8)))  uint32_t v8u;

#define DM 2048   // d_model
#define CL 64     // code length
#define NC 128    // number of codewords
#define WAVES 8
#define RPB 128   // rows per block (8 waves * 16 rows)

// LDS image element offsets (kernel 1)
#define OFF_WTO 0        // 131072 elems: W_to  [64][2048] swz
#define OFF_CW  131072   //   8192 elems: cw    [128][64]  swz
#define OFF_CWT 139264   //   8192 elems: cw^T  [64][128]  swz
#define OFF_SCR 147456   //  16384 elems: per-wave scratch
#define IMG1_ELEMS 147456
#define IMG2_ELEMS 131072

__device__ __forceinline__ __bf16 f2bf(float f) { return (__bf16)f; }

__device__ __forceinline__ v16bf mkfrag(v8bf lo, v8bf hi) {
  return __builtin_shufflevector(lo, hi, 0,1,2,3,4,5,6,7,8,9,10,11,12,13,14,15);
}

// load 8 consecutive f32 and convert to bf16
__device__ __forceinline__ v8bf ld8f_bf(const float* __restrict__ p) {
  float4 a = *(const float4*)p;
  float4 b = *(const float4*)(p + 4);
  v8bf r;
  r[0]=f2bf(a.x); r[1]=f2bf(a.y); r[2]=f2bf(a.z); r[3]=f2bf(a.w);
  r[4]=f2bf(b.x); r[5]=f2bf(b.y); r[6]=f2bf(b.z); r[7]=f2bf(b.w);
  return r;
}

// 16-byte granule (8 bf16) load/store (works for LDS and global pointers)
__device__ __forceinline__ v8bf ldsg(const __bf16* p, int g) {
  return *(const v8bf*)(p + g*8);
}
__device__ __forceinline__ void stsg(__bf16* p, int g, v8bf v) {
  *(v8bf*)(p + g*8) = v;
}

__device__ __forceinline__ v8f wmma_bf16(v16bf a, v16bf b, v8f c) {
  return __builtin_amdgcn_wmma_f32_16x16x32_bf16(false, a, false, b, (short)0, c,
                                                 false, false);
}

// ---- Tensor Data Mover: DMA `rows` x 8192 bf16 (rows*16KB) global -> LDS ----
// D# per CDNA5 ISA (2D tensor, data_size=2B).
__device__ __forceinline__ void tdm_load_lds(uint32_t lds_off, const void* gptr,
                                             uint32_t rows) {
  uint64_t ga = (uint64_t)(uintptr_t)gptr;
  v4u g0;
  g0[0] = 1u;                                   // count=1 (valid descriptor)
  g0[1] = lds_off;                              // lds_addr (bytes)
  g0[2] = (uint32_t)ga;                         // global_addr[31:0]
  g0[3] = (uint32_t)((ga >> 32) & 0x01FFFFFFull) | 0x80000000u; // addr[56:32]|type=2
  v8u g1;
  g1[0] = 0x00010000u;                          // wg_mask=0, data_size=1 (2 bytes)
  g1[1] = 0x20000000u;                          // tensor_dim0=8192 (low16 @ bit48)
  g1[2] = (rows & 0xFFFFu) << 16;               // tensor_dim0 hi16=0 | tensor_dim1
  g1[3] = 0x20000000u;                          // tensor_dim1 hi16=0 | tile_dim0=8192
  g1[4] = rows & 0xFFFFu;                       // tile_dim1=rows, tile_dim2=0
  g1[5] = 8192u;                                // tensor_dim0_stride low32
  g1[6] = 0u;                                   // stride hi16 | dim1_stride lo16
  g1[7] = 0u;
  asm volatile("tensor_load_to_lds %0, %1" :: "s"(g0), "s"(g1) : "memory");
}

// =====================================================================
// Kernel 0 (prep): build swizzled bf16 LDS images of the weights in ws.
// Runs once per launch; ~1.5MB traffic, negligible vs 540MB total.
// =====================================================================
__global__ __launch_bounds__(256)
void k_prep(const float* __restrict__ Wto, const float* __restrict__ Wfrom,
            const float* __restrict__ cw, __bf16* __restrict__ img1,
            __bf16* __restrict__ img2) {
  const int gid = blockIdx.x * 256 + threadIdx.x;   // 16384 threads
  { // W_to: 16384 granules, [l][d] rows of 256 granules, swz mask 15
    int n = gid >> 8, kc = gid & 255;
    stsg(img1 + OFF_WTO, n * 256 + (kc ^ (n & 15)), ld8f_bf(Wto + (size_t)gid * 8));
  }
  { // W_from: 16384 granules, [d][l] rows of 8 granules, swz mask 7
    int d = gid >> 3, kc = gid & 7;
    stsg(img2, d * 8 + (kc ^ (d & 7)), ld8f_bf(Wfrom + (size_t)gid * 8));
  }
  if (gid < 1024) { // cw: [c][d] rows of 8 granules, swz mask 7
    int c = gid >> 3, kc = gid & 7;
    stsg(img1 + OFF_CW, c * 8 + (kc ^ (c & 7)), ld8f_bf(cw + (size_t)gid * 8));
  }
  if (gid < 8192) { // cw^T: [d][c] rows of 16 granules, swz mask 15 (elementwise)
    int d = gid >> 7, c = gid & 127;
    img1[OFF_CWT + (d * 16 + ((c >> 3) ^ (d & 15))) * 8 + (c & 7)] =
        f2bf(cw[c * 64 + d]);
  }
}

// =====================================================================
// Kernel 1: z = x@W_to^T ; soft-quantize vs 128 RM codewords ; q -> ws
// =====================================================================
__global__ __launch_bounds__(256)
void k_quant(const float* __restrict__ x, const __bf16* __restrict__ img1,
             const float* __restrict__ log_temp, float* __restrict__ q,
             int nrows) {
  __shared__ __align__(16) __bf16 smem[163840];     // 320 KB exactly
  __bf16* sWto = smem + OFF_WTO;
  __bf16* sCw  = smem + OFF_CW;
  __bf16* sCwT = smem + OFF_CWT;
  __bf16* sScr = smem + OFF_SCR;

  const int tid  = threadIdx.x;
  const int lane = tid & 31;
  const int wave = tid >> 5;
  const int m    = lane & 15;
  const int h    = lane >> 4;
  const int rowbase = blockIdx.x * RPB + wave * 16;

  // ---- Stage A: one TDM DMA pulls all 288KB of constants into LDS
  if (tid < 32) {
    tdm_load_lds((uint32_t)(uintptr_t)&smem[0], img1, IMG1_ELEMS / 8192);
    __builtin_amdgcn_s_wait_tensorcnt(0);
  }
  __syncthreads();

  // ---- Stage B: z[16x64] = x_tile[16x2048] @ W_to^T  (64 k-steps, 4 N-tiles)
  const float* xrow = x + (size_t)(rowbase + m) * DM;
  v8f zacc[4];
  #pragma unroll
  for (int f = 0; f < 4; ++f) zacc[f] = {};
  for (int kb = 0; kb < DM; kb += 32) {
    v16bf a = mkfrag(ld8f_bf(xrow + kb + 8 * h), ld8f_bf(xrow + kb + 16 + 8 * h));
    const int kc0 = (kb >> 3) + h, kc1 = kc0 + 2;
    #pragma unroll
    for (int f = 0; f < 4; ++f) {
      int n = f * 16 + m;
      v16bf b = mkfrag(ldsg(sWto, n * 256 + (kc0 ^ (n & 15))),
                       ldsg(sWto, n * 256 + (kc1 ^ (n & 15))));
      zacc[f] = wmma_bf16(a, b, zacc[f]);
    }
  }

  // ---- Stage C1: per-row ||z||^2 ; dump z (bf16) to per-wave scratch
  __bf16* zbuf = sScr + wave * 2048;
  float zn[8];
  #pragma unroll
  for (int r = 0; r < 8; ++r) {
    float p = 0.f;
    const int M = r + 8 * h;
    #pragma unroll
    for (int f = 0; f < 4; ++f) {
      float v = zacc[f][r];
      p += v * v;
      int c = m + 16 * f;
      zbuf[(M * 8 + ((c >> 3) ^ (M & 7))) * 8 + (c & 7)] = f2bf(v);
    }
    p += __shfl_xor(p, 1, 32); p += __shfl_xor(p, 2, 32);
    p += __shfl_xor(p, 4, 32); p += __shfl_xor(p, 8, 32);
    zn[r] = p;
  }

  // ---- Stage C2: s[16x128] = z @ cw^T  (2 k-steps, 8 N-tiles)
  v16bf za[2];
  #pragma unroll
  for (int ks = 0; ks < 2; ++ks) {
    int kc0 = ks * 4 + h, kc1 = kc0 + 2;
    za[ks] = mkfrag(ldsg(zbuf, m * 8 + (kc0 ^ (m & 7))),
                    ldsg(zbuf, m * 8 + (kc1 ^ (m & 7))));
  }
  v8f sacc[8];
  #pragma unroll
  for (int f = 0; f < 8; ++f) sacc[f] = {};
  #pragma unroll
  for (int f = 0; f < 8; ++f) {
    int n = f * 16 + m;
    #pragma unroll
    for (int ks = 0; ks < 2; ++ks) {
      int kc0 = ks * 4 + h, kc1 = kc0 + 2;
      v16bf b = mkfrag(ldsg(sCw, n * 8 + (kc0 ^ (n & 7))),
                       ldsg(sCw, n * 8 + (kc1 ^ (n & 7))));
      sacc[f] = wmma_bf16(za[ks], b, sacc[f]);
    }
  }

  // ---- Stage C3: dists + softmax over 128 codes; ||c||^2 == 64 exactly
  float lt   = log_temp[0];
  float temp = fminf(fmaxf(__expf(lt), 0.01f), 5.0f);
  float it   = 1.0f / temp;
  #pragma unroll
  for (int r = 0; r < 8; ++r) {
    const int M = r + 8 * h;
    float lf[8]; float mx = -3.0e30f;
    #pragma unroll
    for (int f = 0; f < 8; ++f) {
      float d2 = zn[r] + 64.0f - 2.0f * sacc[f][r];
      lf[f] = -sqrtf(fmaxf(d2, 0.0f)) * it;
      mx = fmaxf(mx, lf[f]);
    }
    mx = fmaxf(mx, __shfl_xor(mx, 1, 32)); mx = fmaxf(mx, __shfl_xor(mx, 2, 32));
    mx = fmaxf(mx, __shfl_xor(mx, 4, 32)); mx = fmaxf(mx, __shfl_xor(mx, 8, 32));
    float sum = 0.f;
    #pragma unroll
    for (int f = 0; f < 8; ++f) { lf[f] = __expf(lf[f] - mx); sum += lf[f]; }
    sum += __shfl_xor(sum, 1, 32); sum += __shfl_xor(sum, 2, 32);
    sum += __shfl_xor(sum, 4, 32); sum += __shfl_xor(sum, 8, 32);
    float inv = 1.0f / sum;
    #pragma unroll
    for (int f = 0; f < 8; ++f) {
      int c = m + 16 * f;
      zbuf[(M * 16 + ((c >> 3) ^ (M & 15))) * 8 + (c & 7)] = f2bf(lf[f] * inv);
    }
  }

  // ---- Stage C4: q[16x64] = w[16x128] @ cw  (4 k-steps, 4 N-tiles)
  v8f qacc[4];
  #pragma unroll
  for (int f = 0; f < 4; ++f) qacc[f] = {};
  #pragma unroll
  for (int ks = 0; ks < 4; ++ks) {
    int kc0 = ks * 4 + h, kc1 = kc0 + 2;
    v16bf a = mkfrag(ldsg(zbuf, m * 16 + (kc0 ^ (m & 15))),
                     ldsg(zbuf, m * 16 + (kc1 ^ (m & 15))));
    #pragma unroll
    for (int f = 0; f < 4; ++f) {
      int n = f * 16 + m;
      v16bf b = mkfrag(ldsg(sCwT, n * 16 + (kc0 ^ (n & 15))),
                       ldsg(sCwT, n * 16 + (kc1 ^ (n & 15))));
      qacc[f] = wmma_bf16(a, b, qacc[f]);
    }
  }

  #pragma unroll
  for (int r = 0; r < 8; ++r) {
    size_t row = (size_t)(rowbase + r + 8 * h);
    #pragma unroll
    for (int f = 0; f < 4; ++f) q[row * CL + m + 16 * f] = qacc[f][r];
  }
}

// =====================================================================
// Kernel 2: out = x + scale * q @ W_from^T
// =====================================================================
__global__ __launch_bounds__(256)
void k_proj(const float* __restrict__ x, const __bf16* __restrict__ img2,
            const float* __restrict__ q, const float* __restrict__ scale_p,
            float* __restrict__ out, int nrows) {
  __shared__ __align__(16) __bf16 sWf[IMG2_ELEMS];  // 256 KB

  const int tid  = threadIdx.x;
  const int lane = tid & 31;
  const int wave = tid >> 5;
  const int m    = lane & 15;
  const int h    = lane >> 4;
  const int rowbase = blockIdx.x * RPB + wave * 16;

  if (tid < 32) {
    tdm_load_lds((uint32_t)(uintptr_t)&sWf[0], img2, IMG2_ELEMS / 8192);
    __builtin_amdgcn_s_wait_tensorcnt(0);
  }
  __syncthreads();

  const float sc = scale_p[0];

  const float* qrow = q + (size_t)(rowbase + m) * CL;
  v16bf a[2];
  #pragma unroll
  for (int ks = 0; ks < 2; ++ks)
    a[ks] = mkfrag(ld8f_bf(qrow + ks * 32 + 8 * h),
                   ld8f_bf(qrow + ks * 32 + 16 + 8 * h));

  for (int t = 0; t < DM / 16; ++t) {
    v8f acc = {};
    #pragma unroll
    for (int ks = 0; ks < 2; ++ks) {
      int n = t * 16 + m;
      int kc0 = ks * 4 + h, kc1 = kc0 + 2;
      v16bf b = mkfrag(ldsg(sWf, n * 8 + (kc0 ^ (n & 7))),
                       ldsg(sWf, n * 8 + (kc1 ^ (n & 7))));
      acc = wmma_bf16(a[ks], b, acc);
    }
    #pragma unroll
    for (int r = 0; r < 8; ++r) {
      size_t off = (size_t)(rowbase + r + 8 * h) * DM + t * 16 + m;
      out[off] = x[off] + sc * acc[r];
    }
  }
}

// =====================================================================
extern "C" void kernel_launch(void* const* d_in, const int* in_sizes, int n_in,
                              void* d_out, int out_size, void* d_ws, size_t ws_size,
                              hipStream_t stream) {
  const float* x     = (const float*)d_in[0];
  const float* Wto   = (const float*)d_in[1];
  const float* Wfrom = (const float*)d_in[2];
  const float* lt    = (const float*)d_in[3];
  const float* sc    = (const float*)d_in[4];
  const float* cw    = (const float*)d_in[5];
  float*       outp  = (float*)d_out;

  const int nrows  = in_sizes[0] / DM;          // 16384
  const int blocks = nrows / RPB;               // 128

  // workspace layout: q (f32) | img1 (bf16) | img2 (bf16)
  float*  qws  = (float*)d_ws;
  __bf16* img1 = (__bf16*)((char*)d_ws + (size_t)nrows * CL * sizeof(float));
  __bf16* img2 = img1 + IMG1_ELEMS;

  hipLaunchKernelGGL(k_prep, dim3(64), dim3(256), 0, stream, Wto, Wfrom, cw,
                     img1, img2);
  hipLaunchKernelGGL(k_quant, dim3(blocks), dim3(256), 0, stream,
                     x, img1, lt, qws, nrows);
  hipLaunchKernelGGL(k_proj, dim3(blocks), dim3(256), 0, stream,
                     x, img2, qws, sc, outp, nrows);
}
